// CGVAEDecoder_56899726737677
// MI455X (gfx1250) — compile-verified, matrix operand order checked by reading, not run on hardware
//
#include <hip/hip_runtime.h>
#include <hip/hip_bf16.h>

typedef __attribute__((ext_vector_type(16))) _Float16 v16h;
typedef __attribute__((ext_vector_type(8)))  _Float16 v8h;
typedef __attribute__((ext_vector_type(8)))  float    v8f;

#define PV   65536
#define PG   2048
#define NPG  32
#define ED   64
#define HID  64
#define NLAY 12
#define NET  3
#define EPT  131072
#define VEPG 16
#define VE   (PG * VEPG)          // 32768
#define EFD  3
#define GLB  128
#define HEADS 4
#define HDIM (GLB / HEADS)        // 32
#define DTRUNC 10
#define NREPR (HID * (1 + NLAY))  // 832
#define CDIM  (GLB + 2 * NREPR + EFD) // 1795
#define CDIMP 1824                // CDIM padded to multiple of 32
#define ROWS  (VE + PG)           // 34816

// ---------------------------------------------------------------------------
// Pack a KxN row-major f32 weight matrix into the WMMA f16 B-fragment layout:
//   Bp[((ntile*ksteps + step)*32 + lane)*16 + e]
// zero-padded for kk >= K or n >= N. e = element of v16h for that lane.
// ---------------------------------------------------------------------------
__global__ void k_pack_B(const float* __restrict__ B, int ldb, int K, int N,
                         int ksteps, int ntiles, _Float16* __restrict__ Bp) {
  int i = blockIdx.x * blockDim.x + threadIdx.x;
  int total = ntiles * ksteps * 512;
  if (i >= total) return;
  int e    = i & 15;
  int lane = (i >> 4) & 31;
  int idx  = i >> 9;
  int step = idx % ksteps;
  int ntile = idx / ksteps;
  int khalf = (lane >> 4) * 8;
  int j = e >> 1;
  int kk = step * 32 + ((j < 4) ? (khalf + 2 * j) : (16 + khalf + 2 * (j - 4))) + (e & 1);
  int n = ntile * 16 + (lane & 15);
  float v = (kk < K && n < N) ? B[(size_t)kk * ldb + n] : 0.f;
  Bp[i] = (_Float16)v;
}

// ---------------------------------------------------------------------------
// Wave-level WMMA GEMM, f16 A (row-major, lda in halves), packed f16 B.
// Each wave: one 16-row tile x NT 16-col tiles. Block = 4 waves = 4 m-tiles.
// Requires M % 64 == 0, K % 32 == 0 (pre-padded), 16B-aligned rows.
// ACT: 0 none, 1 relu, 2 lrelu(0.2).  CT: float or _Float16 output.
// ---------------------------------------------------------------------------
template <int NT, int ACT, bool ACCUM, bool BIAS, typename CT>
__global__ __launch_bounds__(128)
void gemm_f16(const _Float16* __restrict__ A, int lda,
              const _Float16* __restrict__ Bp, int ksteps,
              const float* __restrict__ bias,
              CT* __restrict__ C, int ldc, int N) {
  const int lane   = threadIdx.x & 31;
  const int wave   = threadIdx.x >> 5;
  const int tile_m = (blockIdx.y * 4 + wave) * 16;
  const int ntile0 = blockIdx.x * NT;
  const int m      = tile_m + (lane & 15);
  const int khalf  = (lane >> 4) * 8;

  const _Float16* __restrict__ arow = A + (size_t)m * lda + khalf;
  const _Float16* __restrict__ bbase = Bp + ((size_t)ntile0 * ksteps * 32 + lane) * 16;

  v8f acc[NT];
  #pragma unroll
  for (int t = 0; t < NT; ++t) acc[t] = (v8f){};

  for (int s = 0; s < ksteps; ++s) {
    const v8h lo = *(const v8h*)(arow + s * 32);
    const v8h hi = *(const v8h*)(arow + s * 32 + 16);
    v16h a;
    #pragma unroll
    for (int i = 0; i < 8; ++i) { a[i] = lo[i]; a[8 + i] = hi[i]; }
    #pragma unroll
    for (int t = 0; t < NT; ++t) {
      const v16h b = *(const v16h*)(bbase + ((size_t)t * ksteps + s) * 512);
      acc[t] = __builtin_amdgcn_wmma_f32_16x16x32_f16(
          false, a, false, b, (short)0, acc[t], false, false);
    }
  }

  // D layout: lane col = (lane&15); VGPR r -> row tile_m + r + (lane>=16 ? 8 : 0)
  const int rbase = tile_m + ((lane >> 4) << 3);
  #pragma unroll
  for (int t = 0; t < NT; ++t) {
    const int colN = (ntile0 + t) * 16 + (lane & 15);
    if (colN >= N) continue;
    const float bv = BIAS ? bias[colN] : 0.f;
    #pragma unroll
    for (int r = 0; r < 8; ++r) {
      float v = acc[t][r] + bv;
      if (ACT == 1) v = fmaxf(v, 0.f);
      if (ACT == 2) v = (v > 0.f) ? v : 0.2f * v;
      const size_t idx = (size_t)(rbase + r) * ldc + colN;
      if (ACCUM) C[idx] += (CT)v; else C[idx] = (CT)v;
    }
  }
}

// elementwise f32 -> f16
__global__ void k_cvt_h(const float* __restrict__ src, _Float16* __restrict__ dst,
                        int n) {
  int i = blockIdx.x * blockDim.x + threadIdx.x;
  if (i < n) dst[i] = (_Float16)src[i];
}

// h[focus[g]] += W_init[ED] (focus-bit row of W_init)
__global__ void k_add_focus(float* __restrict__ h, const int* __restrict__ focus,
                            const float* __restrict__ W_init) {
  int i = blockIdx.x * blockDim.x + threadIdx.x;
  if (i >= PG * HID) return;
  int g = i / HID, j = i % HID;
  h[(size_t)focus[g] * HID + j] += W_init[(size_t)ED * HID + j];
}

// agg[dst] += msg[src]; one thread per (edge, 4 features)
__global__ void k_scatter(const int* __restrict__ adj,
                          const float* __restrict__ msg,
                          float* __restrict__ agg) {
  int i = blockIdx.x * blockDim.x + threadIdx.x;
  if (i >= EPT * (HID / 4)) return;
  int e = i >> 4, j = (i & 15) * 4;
  int src = adj[2 * e], dst = adj[2 * e + 1];
  const float4 v = *(const float4*)(msg + (size_t)src * HID + j);
  float* p = agg + (size_t)dst * HID + j;
  atomicAdd(p + 0, v.x);
  atomicAdd(p + 1, v.y);
  atomicAdd(p + 2, v.z);
  atomicAdd(p + 3, v.w);
}

// append h (f32, PVxHID) into node_repr (f16, PVxNREPR) at column c0
__global__ void k_append_repr(const float* __restrict__ h,
                              _Float16* __restrict__ nrep, int c0) {
  int i = blockIdx.x * blockDim.x + threadIdx.x;
  if (i >= PV * HID) return;
  int r = i / HID, c = i % HID;
  nrep[(size_t)r * NREPR + c0 + c] = (_Float16)h[i];
}

// per-graph attention weights: one wave32 per graph (nodes g*32 .. g*32+31)
__global__ __launch_bounds__(32)
void k_softmax_w(const float* __restrict__ scores, float* __restrict__ w) {
  int g = blockIdx.x, lane = threadIdx.x;
  int node = g * NPG + lane;
  #pragma unroll
  for (int hh = 0; hh < HEADS; ++hh) {
    float v = scores[(size_t)node * HEADS + hh];
    float mx = v;
    for (int off = 16; off; off >>= 1) mx = fmaxf(mx, __shfl_xor(mx, off, 32));
    float ex = __expf(v - mx);
    float sm = ex;
    for (int off = 16; off; off >>= 1) sm += __shfl_xor(sm, off, 32);
    w[(size_t)node * HEADS + hh] = ex / sm;
  }
}

// graph_repr[g, f] = sum_{i<32} trans[g*32+i, f] * w[g*32+i, f/HDIM]
__global__ __launch_bounds__(128)
void k_graph_repr(const float* __restrict__ trans, const float* __restrict__ w,
                  float* __restrict__ grepr) {
  int g = blockIdx.x, f = threadIdx.x;
  int head = f / HDIM;
  float s = 0.f;
  for (int i = 0; i < NPG; ++i) {
    int node = g * NPG + i;
    s += trans[(size_t)node * GLB + f] * w[(size_t)node * HEADS + head];
  }
  grepr[(size_t)g * GLB + f] = s;
}

// edge feature triple: [dist_emb[min(int(ef0),9)], ef1, ef2]
__global__ void k_edge_feat(const float* __restrict__ ef,
                            const float* __restrict__ demb,
                            float* __restrict__ ef3) {
  int e = blockIdx.x * blockDim.x + threadIdx.x;
  if (e >= VE) return;
  int d = (int)ef[(size_t)e * EFD];
  d = d < 0 ? 0 : (d > DTRUNC - 1 ? DTRUNC - 1 : d);
  ef3[(size_t)e * EFD]     = demb[d];
  ef3[(size_t)e * EFD + 1] = ef[(size_t)e * EFD + 1];
  ef3[(size_t)e * EFD + 2] = ef[(size_t)e * EFD + 2];
}

// assemble feats (ROWS x CDIMP, f16), zero-padded columns [CDIM, CDIMP)
__global__ void k_build_feats(_Float16* __restrict__ feats,
                              const float* __restrict__ grepr,
                              const _Float16* __restrict__ nrep,
                              const float* __restrict__ ef3,
                              const int* __restrict__ vec,
                              const int* __restrict__ n2g,
                              const int* __restrict__ focus,
                              const float* __restrict__ stopr) {
  size_t i = (size_t)blockIdx.x * blockDim.x + threadIdx.x;
  if (i >= (size_t)ROWS * CDIMP) return;
  int row = (int)(i / CDIMP), col = (int)(i % CDIMP);
  float v = 0.f;
  if (col < CDIM) {
    if (row < VE) {
      int src = vec[2 * row], tgt = vec[2 * row + 1];
      if (col < GLB) {
        v = grepr[(size_t)n2g[src] * GLB + col];
      } else if (col < GLB + NREPR) {
        v = (float)nrep[(size_t)src * NREPR + (col - GLB)];
      } else if (col < GLB + 2 * NREPR) {
        v = (float)nrep[(size_t)tgt * NREPR + (col - GLB - NREPR)];
      } else {
        v = ef3[(size_t)row * EFD + (col - GLB - 2 * NREPR)];
      }
    } else {
      int g = row - VE;
      if (col < GLB) {
        v = grepr[(size_t)g * GLB + col];
      } else if (col < GLB + NREPR) {
        v = (float)nrep[(size_t)focus[g] * NREPR + (col - GLB)];
      } else {
        v = stopr[col - GLB - NREPR];   // stop_node_repr, 835 wide
      }
    }
  }
  feats[i] = (_Float16)v;
}

// tiny dense output layer: out[r, 0..NOUT) = A[r,:K] @ W + b
template <int NOUT, typename AT>
__global__ void k_dense_small(const AT* __restrict__ A, int lda,
                              const float* __restrict__ W,
                              const float* __restrict__ b,
                              float* __restrict__ out, int ldo, int M, int K) {
  int r = blockIdx.x * blockDim.x + threadIdx.x;
  if (r >= M) return;
  #pragma unroll
  for (int n = 0; n < NOUT; ++n) {
    float s = b[n];
    for (int k = 0; k < K; ++k) s += (float)A[(size_t)r * lda + k] * W[k * NOUT + n];
    out[(size_t)r * ldo + n] = s;
  }
}

// per-graph log-softmax over 16 cand logits + 1 stop logit
__global__ void k_seg_logprob(const float* __restrict__ lg, float* __restrict__ out) {
  int g = blockIdx.x * blockDim.x + threadIdx.x;
  if (g >= PG) return;
  float mx = lg[VE + g];
  for (int i = 0; i < VEPG; ++i) mx = fmaxf(mx, lg[g * VEPG + i]);
  float s = __expf(lg[VE + g] - mx);
  for (int i = 0; i < VEPG; ++i) s += __expf(lg[g * VEPG + i] - mx);
  float lse = __logf(s);
  for (int i = 0; i < VEPG; ++i)
    out[g * VEPG + i] = lg[g * VEPG + i] - mx - lse;
  out[VE + g] = lg[VE + g] - mx - lse;
}

// ---------------------------------------------------------------------------

extern "C" void kernel_launch(void* const* d_in, const int* in_sizes, int n_in,
                              void* d_out, int out_size, void* d_ws, size_t ws_size,
                              hipStream_t stream) {
  const float* node_features = (const float*)d_in[0];
  const int*   adjacency     = (const int*)d_in[1];   // [3, EPT, 2]
  const int*   focus         = (const int*)d_in[3];   // [PG]
  const int*   n2g           = (const int*)d_in[4];   // [PV]
  const int*   vec           = (const int*)d_in[5];   // [VE, 2]
  const float* edge_features = (const float*)d_in[6]; // [VE, 3]
  const float* W_init   = (const float*)d_in[7];      // [65, 64]
  const float* b_init   = (const float*)d_in[8];
  const float* W_msg    = (const float*)d_in[9];      // [12, 4, 64, 64]
  const float* W_dense  = (const float*)d_in[10];     // [3, 64, 64]
  const float* b_dense  = (const float*)d_in[11];     // [3, 64]
  const float* score_W1 = (const float*)d_in[12];
  const float* score_b1 = (const float*)d_in[13];
  const float* score_W2 = (const float*)d_in[14];
  const float* score_b2 = (const float*)d_in[15];
  const float* trans_W1 = (const float*)d_in[16];
  const float* trans_b1 = (const float*)d_in[17];
  const float* trans_W2 = (const float*)d_in[18];
  const float* trans_b2 = (const float*)d_in[19];
  const float* ecs_W1   = (const float*)d_in[20];
  const float* ecs_b1   = (const float*)d_in[21];
  const float* ecs_W2   = (const float*)d_in[22];
  const float* ecs_b2   = (const float*)d_in[23];
  const float* ecs_W3   = (const float*)d_in[24];
  const float* ecs_b3   = (const float*)d_in[25];
  const float* ets_W1   = (const float*)d_in[26];
  const float* ets_b1   = (const float*)d_in[27];
  const float* ets_W2   = (const float*)d_in[28];
  const float* ets_b2   = (const float*)d_in[29];
  const float* stopr    = (const float*)d_in[30];     // [835]
  const float* demb     = (const float*)d_in[31];     // [10]

  float* out = (float*)d_out; // [34816] logprobs ++ [32768*3] type logits

  // ---- workspace carve-out ----
  size_t off = 0;
  char* base = (char*)d_ws;
  auto carve = [&](size_t bytes) -> void* {
    void* p = base + off;
    off = (off + bytes + 255) & ~(size_t)255;
    return p;
  };
  float*    hA     = (float*)carve((size_t)PV * HID * 4);
  float*    hB     = (float*)carve((size_t)PV * HID * 4);
  float*    hC     = (float*)carve((size_t)PV * HID * 4);
  _Float16* nf16   = (_Float16*)carve((size_t)PV * ED * 2);
  _Float16* h16    = (_Float16*)carve((size_t)PV * HID * 2);
  _Float16* nrep   = (_Float16*)carve((size_t)PV * NREPR * 2);
  _Float16* mid128 = (_Float16*)carve((size_t)PV * GLB * 2);
  float*    trans  = (float*)carve((size_t)PV * GLB * 4);
  float*    scores = (float*)carve((size_t)PV * HEADS * 4);
  float*    wgt    = (float*)carve((size_t)PV * HEADS * 4);
  float*    grepr  = (float*)carve((size_t)PG * GLB * 4);
  float*    ef3    = (float*)carve((size_t)VE * EFD * 4);
  _Float16* feats  = (_Float16*)carve((size_t)ROWS * CDIMP * 2);
  _Float16* e1h    = (_Float16*)carve((size_t)ROWS * 64 * 2);
  _Float16* e2h    = (_Float16*)carve((size_t)ROWS * 32 * 2);
  float*    elog   = (float*)carve((size_t)ROWS * 4);
  _Float16* tetsh  = (_Float16*)carve((size_t)VE * 32 * 2);

  // packed weights (halves): [ntile][kstep][lane][16]
  _Float16* wp_init   = (_Float16*)carve((size_t)4 * 2  * 512 * 2);
  _Float16* wp_msg    = (_Float16*)carve((size_t)48 * 4 * 2 * 512 * 2);
  _Float16* wp_dense  = (_Float16*)carve((size_t)3 * 4 * 2 * 512 * 2);
  _Float16* wp_score1 = (_Float16*)carve((size_t)8 * 26 * 512 * 2);
  _Float16* wp_score2 = (_Float16*)carve((size_t)1 * 4  * 512 * 2);
  _Float16* wp_trans1 = (_Float16*)carve((size_t)8 * 26 * 512 * 2);
  _Float16* wp_trans2 = (_Float16*)carve((size_t)8 * 4  * 512 * 2);
  _Float16* wp_ecs1   = (_Float16*)carve((size_t)4 * 57 * 512 * 2);
  _Float16* wp_ecs2   = (_Float16*)carve((size_t)2 * 2  * 512 * 2);
  _Float16* wp_ets1   = (_Float16*)carve((size_t)2 * 57 * 512 * 2);

  auto packB = [&](const float* B, int ldb, int K, int N, int ksteps, int ntiles,
                   _Float16* Bp) {
    int total = ntiles * ksteps * 512;
    k_pack_B<<<(total + 255) / 256, 256, 0, stream>>>(B, ldb, K, N, ksteps, ntiles, Bp);
  };

  // ---- pack all weights into WMMA B-fragment layout ----
  packB(W_init, HID, HID, HID, 2, 4, wp_init);
  for (int i = 0; i < NLAY * 4; ++i)
    packB(W_msg + (size_t)i * HID * HID, HID, HID, HID, 2, 4,
          wp_msg + (size_t)i * 4096);
  for (int i = 0; i < 3; ++i)
    packB(W_dense + (size_t)i * HID * HID, HID, HID, HID, 2, 4,
          wp_dense + (size_t)i * 4096);
  packB(score_W1, 128, NREPR, 128, 26, 8, wp_score1);
  packB(score_W2, HEADS, 128, HEADS, 4, 1, wp_score2);
  packB(trans_W1, 128, NREPR, 128, 26, 8, wp_trans1);
  packB(trans_W2, GLB, 128, GLB, 4, 8, wp_trans2);
  packB(ecs_W1, 64, CDIM, 64, 57, 4, wp_ecs1);
  packB(ecs_W2, 32, 64, 32, 2, 2, wp_ecs2);
  packB(ets_W1, 32, CDIM, 32, 57, 2, wp_ets1);

  // ---- initial embedding: h = [nf, focus] @ W_init + b ----
  k_cvt_h<<<(PV * ED + 255) / 256, 256, 0, stream>>>(node_features, nf16, PV * ED);
  gemm_f16<4, 0, false, true, float><<<dim3(1, PV / 64), 128, 0, stream>>>(
      nf16, ED, wp_init, 2, b_init, hA, HID, HID);
  k_add_focus<<<(PG * HID + 255) / 256, 256, 0, stream>>>(hA, focus, W_init);
  k_append_repr<<<(PV * HID + 255) / 256, 256, 0, stream>>>(hA, nrep, 0);

  // ---- message passing (layer l input = nrep column block l, already f16) ----
  float* h = hA; float* agg = hB; float* tmp = hC;
  int dense_idx = 0;
  for (int l = 0; l < NLAY; ++l) {
    const _Float16* hl = nrep + (size_t)l * HID;   // lda = NREPR
    hipMemsetAsync(agg, 0, (size_t)PV * HID * 4, stream);
    for (int t = 0; t < NET; ++t) {
      gemm_f16<4, 2, false, false, float><<<dim3(1, PV / 64), 128, 0, stream>>>(
          hl, NREPR, wp_msg + (size_t)(l * 4 + t) * 4096, 2, nullptr, tmp, HID, HID);
      k_scatter<<<(EPT * (HID / 4) + 255) / 256, 256, 0, stream>>>(
          adjacency + (size_t)t * EPT * 2, tmp, agg);
    }
    gemm_f16<4, 2, true, false, float><<<dim3(1, PV / 64), 128, 0, stream>>>(
        hl, NREPR, wp_msg + (size_t)(l * 4 + 3) * 4096, 2, nullptr, agg, HID, HID);
    { float* s = h; h = agg; agg = s; }
    if ((l + 1) % 4 == 0) {
      k_cvt_h<<<(PV * HID + 255) / 256, 256, 0, stream>>>(h, h16, PV * HID);
      gemm_f16<4, 1, false, true, float><<<dim3(1, PV / 64), 128, 0, stream>>>(
          h16, HID, wp_dense + (size_t)dense_idx * 4096, 2,
          b_dense + (size_t)dense_idx * HID, tmp, HID, HID);
      ++dense_idx;
      { float* s = h; h = tmp; tmp = s; }
    }
    k_append_repr<<<(PV * HID + 255) / 256, 256, 0, stream>>>(h, nrep, (l + 1) * HID);
  }

  // ---- score head: relu(nr @ W1 + b1) @ W2 + b2 -> per-graph softmax ----
  gemm_f16<4, 1, false, true, _Float16><<<dim3(2, PV / 64), 128, 0, stream>>>(
      nrep, NREPR, wp_score1, 26, score_b1, mid128, 128, 128);
  gemm_f16<1, 0, false, true, float><<<dim3(1, PV / 64), 128, 0, stream>>>(
      mid128, 128, wp_score2, 4, score_b2, scores, HEADS, HEADS);
  k_softmax_w<<<PG, 32, 0, stream>>>(scores, wgt);

  // ---- trans head + weighted pooling ----
  gemm_f16<4, 1, false, true, _Float16><<<dim3(2, PV / 64), 128, 0, stream>>>(
      nrep, NREPR, wp_trans1, 26, trans_b1, mid128, 128, 128);
  gemm_f16<4, 0, false, true, float><<<dim3(2, PV / 64), 128, 0, stream>>>(
      mid128, 128, wp_trans2, 4, trans_b2, trans, GLB, GLB);
  k_graph_repr<<<PG, 128, 0, stream>>>(trans, wgt, grepr);

  // ---- candidate / stop features (f16, K padded to 1824) ----
  k_edge_feat<<<(VE + 255) / 256, 256, 0, stream>>>(edge_features, demb, ef3);
  {
    size_t tot = (size_t)ROWS * CDIMP;
    k_build_feats<<<(unsigned)((tot + 255) / 256), 256, 0, stream>>>(
        feats, grepr, nrep, ef3, vec, n2g, focus, stopr);
  }

  // ---- ecs MLP: 1795 -> 64 -> 32 -> 1 ----
  gemm_f16<4, 1, false, true, _Float16><<<dim3(1, ROWS / 64), 128, 0, stream>>>(
      feats, CDIMP, wp_ecs1, 57, ecs_b1, e1h, 64, 64);
  gemm_f16<2, 1, false, true, _Float16><<<dim3(1, ROWS / 64), 128, 0, stream>>>(
      e1h, 64, wp_ecs2, 2, ecs_b2, e2h, 32, 32);
  k_dense_small<1, _Float16><<<(ROWS + 255) / 256, 256, 0, stream>>>(
      e2h, 32, ecs_W3, ecs_b3, elog, 1, ROWS, 32);

  // ---- per-graph log-softmax over {16 cand, 1 stop} ----
  k_seg_logprob<<<(PG + 255) / 256, 256, 0, stream>>>(elog, out);

  // ---- ets MLP on cand rows: 1795 -> 32 -> 3 ----
  gemm_f16<2, 1, false, true, _Float16><<<dim3(1, VE / 64), 128, 0, stream>>>(
      feats, CDIMP, wp_ets1, 57, ets_b1, tetsh, 32, 32);
  k_dense_small<3, _Float16><<<(VE + 255) / 256, 256, 0, stream>>>(
      tetsh, 32, ets_W2, ets_b2, out + ROWS, 3, VE, 32);

  (void)in_sizes; (void)n_in; (void)out_size; (void)ws_size;
}
